// GATEncoder_88003879895634
// MI455X (gfx1250) — matrix operand order repaired and checked by reading, not compiled, not run on hardware
//
#include <hip/hip_runtime.h>
#include <hip/hip_fp16.h>

typedef _Float16 h8  __attribute__((ext_vector_type(8)));
typedef _Float16 h16 __attribute__((ext_vector_type(16)));
typedef float    f8  __attribute__((ext_vector_type(8)));

#define NODES 50000
#define FIN   128
#define HID   64
#define HEADS 4
#define OUTC  32
#define D1    (HEADS * HID)   // 256
#define NEG_SLOPE 0.2f

// ---------------------------------------------------------------------------
// Utility kernels
// ---------------------------------------------------------------------------
__global__ void fill_f32(float* __restrict__ p, float v, long n) {
  long i = (long)blockIdx.x * blockDim.x + threadIdx.x;
  if (i < n) p[i] = v;
}

__global__ void to_f16(const float* __restrict__ a, _Float16* __restrict__ b, long n) {
  long i = (long)blockIdx.x * blockDim.x + threadIdx.x;
  if (i < n) b[i] = (_Float16)a[i];
}

// W [K x M] f32  ->  Wt [M x K] f16
__global__ void transpose_to_f16(const float* __restrict__ W, _Float16* __restrict__ Wt,
                                 int K, int M) {
  int i = blockIdx.x * blockDim.x + threadIdx.x;
  if (i >= K * M) return;
  int k = i / M, m = i - k * M;
  Wt[(long)m * K + k] = (_Float16)W[i];
}

// ---------------------------------------------------------------------------
// WMMA GEMM: C[nrows x NT*16] = A[nrows x K] * Bt^T + bias
//   A   : f16 row-major [nrows x K]
//   Bt  : f16 row-major [NT*16 x K]  (i.e. W transposed)
// One wave computes a full 16-row strip across all NT column tiles.
// ---------------------------------------------------------------------------
template<int K, int NT>
__global__ __launch_bounds__(256) void gemm_wmma(
    const _Float16* __restrict__ A,
    const _Float16* __restrict__ Bt,
    const float*    __restrict__ bias,
    float*          __restrict__ Cout,
    int nrows)
{
  const int wave  = threadIdx.x >> 5;
  const int lane  = threadIdx.x & 31;
  const int strip = blockIdx.x * 8 + wave;
  if (strip * 16 >= nrows) return;          // wave-uniform: EXEC stays all-ones
  const int rowBase = strip * 16;
  const int M = NT * 16;

  f8 acc[NT];
#pragma unroll
  for (int t = 0; t < NT; ++t)
#pragma unroll
    for (int j = 0; j < 8; ++j) acc[t][j] = 0.0f;

  const int asel = lane >> 4;                       // 0 or 1 (lane half)
  const int arow = rowBase + (lane & 15);           // A row for this lane
  const _Float16* aptr = A + (long)arow * K + asel * 8;
  const int bcol = lane & 15;                       // B column for this lane
  const int bofs = asel * 8;                        // K {0..7,16..23} / {8..15,24..31}

  for (int k0 = 0; k0 < K; k0 += 32) {
    // A tile (16x32 f16): lane<16 -> K {0..7,16..23}, lane>=16 -> K {8..15,24..31}
    h8 alo = *(const h8*)(aptr + k0);
    h8 ahi = *(const h8*)(aptr + k0 + 16);
    h16 a = __builtin_shufflevector(alo, ahi, 0,1,2,3,4,5,6,7,8,9,10,11,12,13,14,15);
#pragma unroll
    for (int t = 0; t < NT; ++t) {
      // B tile (32x16 f16): lane = column, K {0..7,16..23} (lanes 0-15) / {8..15,24..31} (lanes 16-31)
      const _Float16* bp = Bt + (long)(t * 16 + bcol) * K + k0 + bofs;
      h8 blo = *(const h8*)(bp);
      h8 bhi = *(const h8*)(bp + 16);
      h16 b = __builtin_shufflevector(blo, bhi, 0,1,2,3,4,5,6,7,8,9,10,11,12,13,14,15);
      acc[t] = __builtin_amdgcn_wmma_f32_16x16x32_f16(
          false, a, false, b, (short)0, acc[t], false, false);
    }
  }

  // D layout: lane<16 -> VGPR j = C[row j][col lane]; lane>=16 -> C[row 8+j][col lane-16]
  const int colInTile = lane & 15;
  const int rofs = (lane >> 4) * 8;
#pragma unroll
  for (int t = 0; t < NT; ++t) {
    const int col = t * 16 + colInTile;
    const float bv = bias[col];
#pragma unroll
    for (int j = 0; j < 8; ++j) {
      const int r = rowBase + rofs + j;
      Cout[(long)r * M + col] = acc[t][j] + bv;
    }
  }
}

// ---------------------------------------------------------------------------
// Edge pass 1: logits + per-destination segment max.
// One wave per edge; lane covers P = H*C/32 consecutive channels.
// ---------------------------------------------------------------------------
template<int H, int C>
__global__ __launch_bounds__(256) void edge_logits_max(
    const int* __restrict__ ei, int E, int Etot,
    const float* __restrict__ xl, const float* __restrict__ xr,
    const float* __restrict__ att,      // [H*C] flat
    float* __restrict__ logit,          // [Etot*H]
    float* __restrict__ mx)             // [NODES*H], init -FLT_MAX
{
  constexpr int D   = H * C;
  constexpr int P   = D / 32;
  constexpr int LPH = 32 / H;           // lanes per head
  const int lane = threadIdx.x & 31;
  const int e = blockIdx.x * 8 + (threadIdx.x >> 5);
  if (e >= Etot) return;
  int s, d;
  if (e < E) { s = ei[e]; d = ei[E + e]; } else { s = e - E; d = s; }

  const float* pl = xl + (long)s * D + lane * P;
  const float* pr = xr + (long)d * D + lane * P;
  float sum = 0.f;
#pragma unroll
  for (int i = 0; i < P; ++i) {
    float v = pl[i] + pr[i];
    v = v > 0.f ? v : NEG_SLOPE * v;
    sum += v * att[lane * P + i];
  }
#pragma unroll
  for (int off = LPH >> 1; off > 0; off >>= 1)
    sum += __shfl_xor(sum, off, 32);

  const int h = (lane * P) / C;
  if ((lane % LPH) == 0) {
    logit[(long)e * H + h] = sum;
    __hip_atomic_fetch_max(&mx[(long)d * H + h], sum,
                           __ATOMIC_RELAXED, __HIP_MEMORY_SCOPE_AGENT);
  }
}

// ---------------------------------------------------------------------------
// Edge pass 2: alpha = exp(logit - mx[dst]); denom[dst] += alpha (in-place)
// ---------------------------------------------------------------------------
__global__ void edge_softmax_denom(
    const int* __restrict__ ei, int E, int Etot, int H,
    float* __restrict__ logit,           // in: logit, out: alpha
    const float* __restrict__ mx,
    float* __restrict__ denom)           // init 0
{
  long i = (long)blockIdx.x * blockDim.x + threadIdx.x;
  if (i >= (long)Etot * H) return;
  int e = (int)(i / H), h = (int)(i - (long)e * H);
  int d = (e < E) ? ei[E + e] : (e - E);
  float a = __expf(logit[i] - mx[(long)d * H + h]);
  logit[i] = a;
  atomicAdd(&denom[(long)d * H + h], a);
}

// ---------------------------------------------------------------------------
// Edge pass 3: out[dst] += (alpha/denom) * xl[src]   (scatter-add)
// ---------------------------------------------------------------------------
template<int H, int C>
__global__ __launch_bounds__(256) void edge_aggregate(
    const int* __restrict__ ei, int E, int Etot,
    const float* __restrict__ xl,
    const float* __restrict__ alpha,     // [Etot*H]
    const float* __restrict__ denom,     // [NODES*H]
    float* __restrict__ out)             // [NODES*D], init 0
{
  constexpr int D = H * C;
  constexpr int P = D / 32;
  const int lane = threadIdx.x & 31;
  const int e = blockIdx.x * 8 + (threadIdx.x >> 5);
  if (e >= Etot) return;
  int s, d;
  if (e < E) { s = ei[e]; d = ei[E + e]; } else { s = e - E; d = s; }

  const int h = (lane * P) / C;
  const float w = alpha[(long)e * H + h] / (denom[(long)d * H + h] + 1e-16f);
  const float* pl = xl + (long)s * D + lane * P;
  float* po = out + (long)d * D + lane * P;
  __builtin_prefetch(pl, 0, 0);
#pragma unroll
  for (int i = 0; i < P; ++i)
    atomicAdd(&po[i], w * pl[i]);
}

// ---------------------------------------------------------------------------
// finalize layer 1: h = elu(agg + bias1) -> f16 (feeds layer-2 GEMM)
// ---------------------------------------------------------------------------
__global__ void finalize1(const float* __restrict__ agg, const float* __restrict__ bias,
                          _Float16* __restrict__ hh, long n, int D) {
  long i = (long)blockIdx.x * blockDim.x + threadIdx.x;
  if (i >= n) return;
  float v = agg[i] + bias[i % D];
  v = v > 0.f ? v : (__expf(v) - 1.f);
  hh[i] = (_Float16)v;
}

// finalize layer 2: d_out = agg2 + bias2  (H=1 so mean over heads is identity)
__global__ void finalize2(const float* __restrict__ agg, const float* __restrict__ bias,
                          float* __restrict__ out, long n, int D) {
  long i = (long)blockIdx.x * blockDim.x + threadIdx.x;
  if (i >= n) return;
  out[i] = agg[i] + bias[i % D];
}

// ---------------------------------------------------------------------------
// Launch
// ---------------------------------------------------------------------------
extern "C" void kernel_launch(void* const* d_in, const int* in_sizes, int n_in,
                              void* d_out, int out_size, void* d_ws, size_t ws_size,
                              hipStream_t stream) {
  const float* x    = (const float*)d_in[0];
  const int*   ei   = (const int*)  d_in[1];
  const float* Wl1  = (const float*)d_in[2];
  const float* bl1  = (const float*)d_in[3];
  const float* Wr1  = (const float*)d_in[4];
  const float* br1  = (const float*)d_in[5];
  const float* att1 = (const float*)d_in[6];
  const float* bias1= (const float*)d_in[7];
  const float* Wl2  = (const float*)d_in[8];
  const float* bl2  = (const float*)d_in[9];
  const float* Wr2  = (const float*)d_in[10];
  const float* br2  = (const float*)d_in[11];
  const float* att2 = (const float*)d_in[12];
  const float* bias2= (const float*)d_in[13];
  float* outp = (float*)d_out;

  const int E    = in_sizes[1] / 2;
  const int Etot = E + NODES;

  // -------- workspace carve --------
  char* w = (char*)d_ws;
  auto carve = [&](size_t bytes) -> void* {
    void* p = (void*)w; w += (bytes + 255) & ~(size_t)255; return p;
  };
  _Float16* xh    = (_Float16*)carve((size_t)NODES * FIN * 2);
  _Float16* wl1t  = (_Float16*)carve((size_t)D1 * FIN * 2);
  _Float16* wr1t  = (_Float16*)carve((size_t)D1 * FIN * 2);
  _Float16* wl2t  = (_Float16*)carve((size_t)OUTC * D1 * 2);
  _Float16* wr2t  = (_Float16*)carve((size_t)OUTC * D1 * 2);
  float* xl1      = (float*)carve((size_t)NODES * D1 * 4);
  float* xr1      = (float*)carve((size_t)NODES * D1 * 4);
  float* agg1     = (float*)carve((size_t)NODES * D1 * 4);
  float* logit1   = (float*)carve((size_t)Etot * HEADS * 4);
  float* mx1      = (float*)carve((size_t)NODES * HEADS * 4);
  float* den1     = (float*)carve((size_t)NODES * HEADS * 4);
  _Float16* hh    = (_Float16*)carve((size_t)NODES * D1 * 2);
  float* xl2      = (float*)carve((size_t)NODES * OUTC * 4);
  float* xr2      = (float*)carve((size_t)NODES * OUTC * 4);
  float* agg2     = (float*)carve((size_t)NODES * OUTC * 4);
  float* logit2   = (float*)carve((size_t)Etot * 4);
  float* mx2      = (float*)carve((size_t)NODES * 4);
  float* den2     = (float*)carve((size_t)NODES * 4);
  (void)ws_size; (void)n_in; (void)out_size;

  const int T = 256;
  auto blocks = [](long n, int t) { return (int)((n + t - 1) / t); };

  // -------- prep: f16 conversions + weight transposes --------
  to_f16<<<blocks((long)NODES * FIN, T), T, 0, stream>>>(x, xh, (long)NODES * FIN);
  transpose_to_f16<<<blocks((long)FIN * D1, T), T, 0, stream>>>(Wl1, wl1t, FIN, D1);
  transpose_to_f16<<<blocks((long)FIN * D1, T), T, 0, stream>>>(Wr1, wr1t, FIN, D1);
  transpose_to_f16<<<blocks((long)D1 * OUTC, T), T, 0, stream>>>(Wl2, wl2t, D1, OUTC);
  transpose_to_f16<<<blocks((long)D1 * OUTC, T), T, 0, stream>>>(Wr2, wr2t, D1, OUTC);

  // -------- init accumulators (ws is poisoned; must re-init every call) ----
  fill_f32<<<blocks((long)NODES * D1, T), T, 0, stream>>>(agg1, 0.f, (long)NODES * D1);
  fill_f32<<<blocks((long)NODES * HEADS, T), T, 0, stream>>>(mx1, -3.402823466e38f, (long)NODES * HEADS);
  fill_f32<<<blocks((long)NODES * HEADS, T), T, 0, stream>>>(den1, 0.f, (long)NODES * HEADS);
  fill_f32<<<blocks((long)NODES * OUTC, T), T, 0, stream>>>(agg2, 0.f, (long)NODES * OUTC);
  fill_f32<<<blocks((long)NODES, T), T, 0, stream>>>(mx2, -3.402823466e38f, (long)NODES);
  fill_f32<<<blocks((long)NODES, T), T, 0, stream>>>(den2, 0.f, (long)NODES);

  const int strips   = NODES / 16;               // 3125 exactly
  const int gemmGrid = (strips + 7) / 8;
  const int edgeGrid = (Etot + 7) / 8;

  // -------- layer 1 --------
  gemm_wmma<FIN, D1 / 16><<<gemmGrid, 256, 0, stream>>>(xh, wl1t, bl1, xl1, NODES);
  gemm_wmma<FIN, D1 / 16><<<gemmGrid, 256, 0, stream>>>(xh, wr1t, br1, xr1, NODES);

  edge_logits_max<HEADS, HID><<<edgeGrid, 256, 0, stream>>>(
      ei, E, Etot, xl1, xr1, att1, logit1, mx1);
  edge_softmax_denom<<<blocks((long)Etot * HEADS, T), T, 0, stream>>>(
      ei, E, Etot, HEADS, logit1, mx1, den1);
  edge_aggregate<HEADS, HID><<<edgeGrid, 256, 0, stream>>>(
      ei, E, Etot, xl1, logit1, den1, agg1);

  finalize1<<<blocks((long)NODES * D1, T), T, 0, stream>>>(
      agg1, bias1, hh, (long)NODES * D1, D1);

  // -------- layer 2 --------
  gemm_wmma<D1, OUTC / 16><<<gemmGrid, 256, 0, stream>>>(hh, wl2t, bl2, xl2, NODES);
  gemm_wmma<D1, OUTC / 16><<<gemmGrid, 256, 0, stream>>>(hh, wr2t, br2, xr2, NODES);

  edge_logits_max<1, OUTC><<<edgeGrid, 256, 0, stream>>>(
      ei, E, Etot, xl2, xr2, att2, logit2, mx2);
  edge_softmax_denom<<<blocks((long)Etot, T), T, 0, stream>>>(
      ei, E, Etot, 1, logit2, mx2, den2);
  edge_aggregate<1, OUTC><<<edgeGrid, 256, 0, stream>>>(
      ei, E, Etot, xl2, logit2, den2, agg2);

  finalize2<<<blocks((long)NODES * OUTC, T), T, 0, stream>>>(
      agg2, bias2, outp, (long)NODES * OUTC, OUTC);
}
